// Attention_5420248728256
// MI455X (gfx1250) — matrix operand order repaired and checked by reading, not compiled, not run on hardware
//
#include <hip/hip_runtime.h>
#include <hip/hip_bf16.h>
#include <math.h>

// FlashAttention-2 forward, causal. B=2, NH=16, T=2048, D=64.
// 4-wave workgroup = 64 query rows of one head; K/V blocks staged once per
// block into double-buffered LDS as f16. f16 WMMA (16x16x32), f32 accum.
// Row-max via single-instruction DPP-fused v_max_num_f32; rowsum via ones WMMA.
// Diagonal (masked) block is a separate template specialization: hot loop has
// zero masking code.

typedef _Float16 f16;
typedef _Float16 v16h __attribute__((ext_vector_type(16)));
typedef _Float16 v8h  __attribute__((ext_vector_type(8)));
typedef _Float16 v4h  __attribute__((ext_vector_type(4)));
typedef float    v8f  __attribute__((ext_vector_type(8)));

#define TDIM  2048
#define DDIM  64
#define QTILE 16
#define KBLK  32
#define WAVES 4
#define QROWS (QTILE * WAVES)   // 64 query rows per block

#define WMMA_F16(a, b, c) \
    __builtin_amdgcn_wmma_f32_16x16x32_f16(false, (a), false, (b), (short)0, (c), false, false)

// r = max(dpp(x), x) in ONE VALU op: DPP modifier applied to src0 of v_max_num_f32.
static __device__ __forceinline__ float maxdpp_xor1(float x) {
    float r;
    asm("v_max_num_f32_dpp %0, %1, %2 dpp8:[1,0,3,2,5,4,7,6]"
        : "=v"(r) : "v"(x), "v"(x));
    return r;
}
static __device__ __forceinline__ float maxdpp_xor2(float x) {
    float r;
    asm("v_max_num_f32_dpp %0, %1, %2 dpp8:[2,3,0,1,6,7,4,5]"
        : "=v"(r) : "v"(x), "v"(x));
    return r;
}
static __device__ __forceinline__ float maxdpp_xor4(float x) {
    float r;
    asm("v_max_num_f32_dpp %0, %1, %2 dpp8:[4,5,6,7,0,1,2,3]"
        : "=v"(r) : "v"(x), "v"(x));
    return r;
}
// xor-8 within a 16-lane row == row_ror:8 (DPP16).
static __device__ __forceinline__ float maxdpp_ror8(float x) {
    float r;
    asm("v_max_num_f32_dpp %0, %1, %2 row_ror:8 row_mask:0xf bank_mask:0xf"
        : "=v"(r) : "v"(x), "v"(x));
    return r;
}
// max across the 16 lanes of each half-wave (rows live per 16-lane half).
static __device__ __forceinline__ float rowmax16(float x) {
    x = maxdpp_xor1(x);
    x = maxdpp_xor2(x);
    x = maxdpp_xor4(x);
    x = maxdpp_ror8(x);
    return x;
}

// 16 consecutive f16 in LDS -> v16h (B-operand element order: K = e).
static __device__ __forceinline__ v16h ldh16(const f16* p) {
    const v8h* q = (const v8h*)p;
    v8h x = q[0], y = q[1];
    v16h r;
    #pragma unroll
    for (int e = 0; e < 8; ++e) { r[e] = x[e]; r[e + 8] = y[e]; }
    return r;
}

// One 32-key block: S = Q K^T, online softmax, O += P V.  DIAG compiles the
// causal-mask bias in (via fma); the non-DIAG hot path carries no mask code.
template <bool DIAG>
static __device__ __forceinline__ void attn_step(
    const v16h (&qa)[2], const v16h& ones,
    const f16* kd, const f16* vd, f16* pl,
    v8f& o0, v8f& o1, v8f& o2, v8f& o3, v8f& o4,
    float (&m)[8], const float (&bias0)[8], const float (&bias1)[8],
    int row16, int hi, int subA, int subB)
{
    const float scale2 = 0.125f * 1.44269504089f;   // 1/sqrt(64) * log2(e)

    // ---- all four K B-operands up front, then 4 chained WMMAs ----
    v16h kb00 = ldh16(kd + (row16)      * DDIM +  0 + subB);
    v16h kb10 = ldh16(kd + (16 + row16) * DDIM +  0 + subB);
    v16h kb01 = ldh16(kd + (row16)      * DDIM + 32 + subB);
    v16h kb11 = ldh16(kd + (16 + row16) * DDIM + 32 + subB);

    v8f s0 = {}, s1 = {};
    s0 = WMMA_F16(qa[0], kb00, s0);
    s1 = WMMA_F16(qa[0], kb10, s1);
    s0 = WMMA_F16(qa[1], kb01, s0);
    s1 = WMMA_F16(qa[1], kb11, s1);

    // ---- online softmax in log2 domain (row stats per 16-lane half) ----
    float p0[8], p1[8];
    #pragma unroll
    for (int r = 0; r < 8; ++r) {
        float a, b;
        if (DIAG) {
            a = fmaf(s0[r], scale2, bias0[r]);   // bias is 0 or -inf (precomputed)
            b = fmaf(s1[r], scale2, bias1[r]);
        } else {
            a = s0[r] * scale2;
            b = s1[r] * scale2;
        }
        const float mn   = fmaxf(m[r], rowmax16(fmaxf(a, b)));
        const float corr = exp2f(m[r] - mn);
        p0[r] = exp2f(a - mn);
        p1[r] = exp2f(b - mn);
        m[r]  = mn;
        o0[r] *= corr; o1[r] *= corr; o2[r] *= corr; o3[r] *= corr; o4[r] *= corr;
    }

    // ---- V B-operands (independent of P region) issued before P round-trip ----
    v16h vb0 = ldh16(vd + ( 0 + row16) * KBLK + subB);
    v16h vb1 = ldh16(vd + (16 + row16) * KBLK + subB);
    v16h vb2 = ldh16(vd + (32 + row16) * KBLK + subB);
    v16h vb3 = ldh16(vd + (48 + row16) * KBLK + subB);

    // ---- P (C-layout) -> per-wave LDS -> A-layout ----
    #pragma unroll
    for (int r = 0; r < 8; ++r) {
        const int prow = r + hi * 8;
        pl[prow * KBLK + row16]      = (f16)p0[r];
        pl[prow * KBLK + 16 + row16] = (f16)p1[r];
    }
    asm volatile("s_wait_dscnt 0" ::: "memory");    // same-wave LDS RAW fence

    v16h pa;
    {
        v8h x = *(const v8h*)(pl + row16 * KBLK + 2 * subA);
        v8h y = *(const v8h*)(pl + row16 * KBLK + 2 * subA + 8);
        #pragma unroll
        for (int e = 0; e < 8; ++e) { pa[e] = x[e]; pa[e+8] = y[e]; }
    }

    // ---- O += P V (4 d-tiles) + rowsum via ones tile ----
    o0 = WMMA_F16(pa, vb0, o0);
    o1 = WMMA_F16(pa, vb1, o1);
    o2 = WMMA_F16(pa, vb2, o2);
    o3 = WMMA_F16(pa, vb3, o3);
    o4 = WMMA_F16(pa, ones, o4);
}

__global__ __launch_bounds__(WAVES * 32)
void fa_fwd_kernel(const float* __restrict__ Q, const float* __restrict__ K,
                   const float* __restrict__ V, float* __restrict__ O)
{
    // Double-buffered staging (f16): K row-major [key][d], V transposed [d][key].
    __shared__ __align__(16) f16 sk[2][KBLK * DDIM];
    __shared__ __align__(16) f16 sv[2][DDIM * KBLK];
    __shared__ __align__(16) f16 sp[WAVES][QTILE * KBLK];  // per-wave P transpose buffer

    const int tid  = threadIdx.x;
    const int wave = tid >> 5;
    const int lane = tid & 31;

    const int bh    = blockIdx.x >> 5;            // 32 blocks per head
    const int qbase = (blockIdx.x & 31) * QROWS;

    const size_t hoff = (size_t)bh * TDIM * DDIM;
    const float* Kb = K + hoff;
    const float* Vb = V + hoff;
    const int    q_lo = qbase + wave * QTILE;
    const float* Qb = Q + hoff + (size_t)q_lo * DDIM;
    float*       Ob = O + hoff + (size_t)q_lo * DDIM;

    const int row16 = lane & 15;
    const int hi    = lane >> 4;
    const int subA  = hi ? 8 : 0;    // A lane-half k base is 2*subA (contiguous k = 16*hi + e)
    const int subB  = hi ? 16 : 0;   // B lane-half k base (contiguous k = 16*hi + e)

    // ---- Q tile as two A operands (d 0..31, 32..63): two contiguous 8-float runs ----
    v16h qa[2];
    #pragma unroll
    for (int c = 0; c < 2; ++c) {
        const float* qr = Qb + row16 * DDIM + c * 32;
        const float4* r0 = (const float4*)(qr + 2 * subA);
        const float4* r1 = (const float4*)(qr + 2 * subA + 8);
        float4 a0 = r0[0], a1 = r0[1], b0 = r1[0], b1 = r1[1];
        const float t0[8] = {a0.x,a0.y,a0.z,a0.w,a1.x,a1.y,a1.z,a1.w};
        const float t1[8] = {b0.x,b0.y,b0.z,b0.w,b1.x,b1.y,b1.z,b1.w};
        #pragma unroll
        for (int e = 0; e < 8; ++e) { qa[c][e] = (f16)t0[e]; qa[c][e+8] = (f16)t1[e]; }
    }

    // Ones B operand: o4 += P * ones -> rowsum(P) replicated in every lane.
    v16h ones;
    #pragma unroll
    for (int e = 0; e < 16; ++e) ones[e] = (f16)1.0f;

    v8f o0 = {}, o1 = {}, o2 = {}, o3 = {}, o4 = {};
    float m[8];
    #pragma unroll
    for (int r = 0; r < 8; ++r) m[r] = -INFINITY;

    const int nkb        = qbase / KBLK + 2;   // block-wide iteration count
    const int nkb_active = q_lo / KBLK + 1;    // this wave's causal range

    // Precomputed diagonal-block mask bias (0 or -inf); applies only to the
    // wave's last active block.
    const int key0d = (nkb_active - 1) * KBLK;
    float bias0[8], bias1[8];
    #pragma unroll
    for (int r = 0; r < 8; ++r) {
        const int grow = q_lo + r + hi * 8;
        bias0[r] = (key0d + row16      > grow) ? -INFINITY : 0.0f;
        bias1[r] = (key0d + 16 + row16 > grow) ? -INFINITY : 0.0f;
    }

    // ---- register-staged pipeline: preload K/V block 0 ----
    float4 kreg[4], vreg[4];
    {
        const float4* ks = (const float4*)Kb;
        const float4* vs = (const float4*)Vb;
        #pragma unroll
        for (int i = 0; i < 4; ++i) { kreg[i] = ks[i*128 + tid]; vreg[i] = vs[i*128 + tid]; }
    }

    f16* pl = sp[wave];

    for (int kb = 0; kb < nkb; ++kb) {
        const int key0 = kb * KBLK;

        // issue next block's global loads early; prefetch two blocks ahead
        float4 kn[4], vn[4];
        const bool have_next = (kb + 1 < nkb);
        if (have_next) {
            const float4* ks = (const float4*)(Kb + (size_t)(key0 + KBLK) * DDIM);
            const float4* vs = (const float4*)(Vb + (size_t)(key0 + KBLK) * DDIM);
            #pragma unroll
            for (int i = 0; i < 4; ++i) { kn[i] = ks[i*128 + tid]; vn[i] = vs[i*128 + tid]; }
            if (kb + 2 < nkb) {
                __builtin_prefetch(Kb + (size_t)(key0 + 2*KBLK) * DDIM + tid * 16, 0, 0);
                __builtin_prefetch(Vb + (size_t)(key0 + 2*KBLK) * DDIM + tid * 16, 0, 0);
            }
        }

        // ---- cooperative stage of current block into LDS buf[kb&1] ----
        f16* kd = sk[kb & 1];
        f16* vd = sv[kb & 1];
        #pragma unroll
        for (int i = 0; i < 4; ++i) {
            const int e4 = i * 128 + tid;              // 512 float4 per 32x64 block
            v4h kp;
            kp[0] = (f16)kreg[i].x; kp[1] = (f16)kreg[i].y;
            kp[2] = (f16)kreg[i].z; kp[3] = (f16)kreg[i].w;
            *(v4h*)(kd + e4 * 4) = kp;                 // K row-major, packed b64 store
            const int key = e4 >> 4;
            const int d4  = (e4 & 15) << 2;
            vd[(d4+0)*KBLK + key] = (f16)vreg[i].x;    // V transposed [d][key]
            vd[(d4+1)*KBLK + key] = (f16)vreg[i].y;
            vd[(d4+2)*KBLK + key] = (f16)vreg[i].z;
            vd[(d4+3)*KBLK + key] = (f16)vreg[i].w;
        }
        __syncthreads();   // buf[kb&1] ready; buf[(kb+1)&1] free next iteration

        if (kb < nkb_active) {            // wave-uniform
            if (kb == nkb_active - 1) {   // wave-uniform: masked diagonal block
                attn_step<true >(qa, ones, kd, vd, pl, o0, o1, o2, o3, o4,
                                 m, bias0, bias1, row16, hi, subA, subB);
            } else {                      // clean hot path, no mask code
                attn_step<false>(qa, ones, kd, vd, pl, o0, o1, o2, o3, o4,
                                 m, bias0, bias1, row16, hi, subA, subB);
            }
        }

        if (have_next) {
            #pragma unroll
            for (int i = 0; i < 4; ++i) { kreg[i] = kn[i]; vreg[i] = vn[i]; }
        }
    }

    // ---- epilogue: normalize by rowsum (o4 identical across columns) and store ----
    #pragma unroll
    for (int r = 0; r < 8; ++r) {
        const float inv  = 1.0f / o4[r];
        const int   orow = r + hi * 8;
        Ob[orow * DDIM +  0 + row16] = o0[r] * inv;
        Ob[orow * DDIM + 16 + row16] = o1[r] * inv;
        Ob[orow * DDIM + 32 + row16] = o2[r] * inv;
        Ob[orow * DDIM + 48 + row16] = o3[r] * inv;
    }
}

extern "C" void kernel_launch(void* const* d_in, const int* in_sizes, int n_in,
                              void* d_out, int out_size, void* d_ws, size_t ws_size,
                              hipStream_t stream) {
    const float* Q = (const float*)d_in[0];
    const float* K = (const float*)d_in[1];
    const float* V = (const float*)d_in[2];
    float* Out = (float*)d_out;

    const int blocks = 2 * 16 * (TDIM / QROWS);   // 1024 blocks, 128 threads (4 waves)
    fa_fwd_kernel<<<blocks, WAVES * 32, 0, stream>>>(Q, K, V, Out);
}
